// EnhancedGeometryAwareModule_86912958202499
// MI455X (gfx1250) — matrix operand order repaired and checked by reading, not compiled, not run on hardware
//
#include <hip/hip_runtime.h>

typedef __attribute__((ext_vector_type(16))) _Float16 v16h;
typedef __attribute__((ext_vector_type(8)))  float    v8f;
typedef int v4i __attribute__((vector_size(16)));

#define Bsz 16384
#define Dd  1024
#define Kk  10
#define Hh  8

#if defined(__has_builtin)
#if __has_builtin(__builtin_amdgcn_global_load_async_to_lds_b128) && \
    __has_builtin(__builtin_amdgcn_s_wait_asynccnt)
#define USE_ASYNC_LDS 1
#endif
#endif
#ifndef USE_ASYNC_LDS
#define USE_ASYNC_LDS 0
#endif

struct HPack { uint4 a, b; };

__device__ __forceinline__ v16h frag2(const _Float16* p0, const _Float16* p1) {
    HPack t; t.a = *(const uint4*)p0; t.b = *(const uint4*)p1;
    return __builtin_bit_cast(v16h, t);
}

#if USE_ASYNC_LDS
typedef v4i __attribute__((address_space(1)))* gp4_t;   // global int4*
typedef v4i __attribute__((address_space(3)))* lp4_t;   // LDS int4*
// Async global->LDS 16B copy (ASYNCcnt-tracked). Low 32 bits of a generic LDS
// address are the LDS byte address on gfx1250 (ISA §10.2 aperture table), so the
// truncating integer cast to the AS(3) pointer is exact.
__device__ __forceinline__ void async_b128(const _Float16* g, _Float16* l) {
    __builtin_amdgcn_global_load_async_to_lds_b128(
        (gp4_t)(uintptr_t)g,
        (lp4_t)(unsigned int)(uintptr_t)l,
        0, 0);
}
#endif

// ---------------- WMMA GEMM: C[M,N] = A[M,K](f16,row) * BT[N,K](f16) + bias; epilogues ----
// EPI 0: outF = acc + bias
// EPI 1: outF = res + resScale*(acc + bias)        (res may alias outF elementwise)
// EPI 2: outH = gelu_exact(acc + bias)  (f16)
template<int EPI>
__global__ __launch_bounds__(256)
void wmma_gemm_epi(const _Float16* __restrict__ A, const _Float16* __restrict__ BT,
                   const float* __restrict__ bias, const float* __restrict__ res,
                   float* __restrict__ outF, _Float16* __restrict__ outH,
                   int M, int N, int Kd, float resScale)
{
    const int bm   = blockIdx.x * 128;
    const int bn   = blockIdx.y * 128;
    const int tid  = threadIdx.x;
    const int wave = tid >> 5;
    const int lane = tid & 31;
    const int wm   = (wave >> 2) * 64;   // 2 waves in M
    const int wn   = (wave & 3) * 32;    // 4 waves in N
    const int l16  = lane & 15;
    const int lhi  = lane >> 4;

    // double-buffered A tile: 128 rows x 64 halves, padded to stride 72 (conflict-free)
    __shared__ _Float16 As[2][128 * 72];

    v8f acc[4][2];
    const v8f vzero = {};
    #pragma unroll
    for (int i = 0; i < 4; ++i)
        #pragma unroll
        for (int j = 0; j < 2; ++j) acc[i][j] = vzero;

    // staging map: thread -> (row, 32-half chunk); 64B per thread per stage
    const int arow = tid >> 1;
    const int acol = (tid & 1) * 32;
    const _Float16* aSrc = A + (size_t)(bm + arow) * Kd + acol;
    const int aoff = arow * 72 + acol;

    // per-lane B fragment rows (weights pre-transposed [N,K]; 16 contiguous K per lane)
    const _Float16* bBase0 = BT + (size_t)(bn + wn +  0 + l16) * Kd + lhi * 16;
    const _Float16* bBase1 = BT + (size_t)(bn + wn + 16 + l16) * Kd + lhi * 16;

#if USE_ASYNC_LDS
    {   // prologue: stage buffer 0
        _Float16* d = &As[0][aoff];
        #pragma unroll
        for (int c = 0; c < 4; ++c) async_b128(aSrc + c * 8, d + c * 8);
        __builtin_amdgcn_s_wait_asynccnt(0);
        __syncthreads();
    }
#endif

    for (int k0 = 0; k0 < Kd; k0 += 64) {
        const int p = (k0 >> 6) & 1;
#if USE_ASYNC_LDS
        if (k0 + 64 < Kd) {   // overlap next stage copy with this stage's WMMAs
            const _Float16* s = aSrc + k0 + 64;
            _Float16* d = &As[p ^ 1][aoff];
            #pragma unroll
            for (int c = 0; c < 4; ++c) async_b128(s + c * 8, d + c * 8);
        }
#else
        __syncthreads();
        {
            const uint4* s = (const uint4*)(aSrc + k0);
            uint4* d = (uint4*)&As[p][aoff];
            d[0] = s[0]; d[1] = s[1]; d[2] = s[2]; d[3] = s[3];
        }
        __syncthreads();
#endif
        if (k0 + 64 < Kd) {
            __builtin_prefetch(bBase0 + k0 + 64, 0, 1);
            __builtin_prefetch(bBase1 + k0 + 64, 0, 1);
        }

        const _Float16* Ab = &As[p][0];
        #pragma unroll
        for (int kk = 0; kk < 64; kk += 32) {
            v16h bf0 = frag2(bBase0 + k0 + kk, bBase0 + k0 + kk + 8);
            v16h bf1 = frag2(bBase1 + k0 + kk, bBase1 + k0 + kk + 8);
            #pragma unroll
            for (int i = 0; i < 4; ++i) {
                // A layout (matches B): lanes0-15 K[0..15]; lanes16-31 K[16..31]
                const _Float16* ap = Ab + (wm + i * 16 + l16) * 72 + kk + lhi * 16;
                v16h af = frag2(ap, ap + 8);
                acc[i][0] = __builtin_amdgcn_wmma_f32_16x16x32_f16(false, af, false, bf0,
                             (short)0, acc[i][0], false, false);
                acc[i][1] = __builtin_amdgcn_wmma_f32_16x16x32_f16(false, af, false, bf1,
                             (short)0, acc[i][1], false, false);
            }
        }
#if USE_ASYNC_LDS
        __builtin_amdgcn_s_wait_asynccnt(0);   // my next-stage copies landed
        __syncthreads();                        // everyone's landed; reads of p done
#endif
    }

    // epilogue: C layout — VGPR r: lanes0-15 M=r, lanes16-31 M=r+8; N = l16
    #pragma unroll
    for (int i = 0; i < 4; ++i) {
        const int mBase = bm + wm + i * 16 + lhi * 8;
        #pragma unroll
        for (int j = 0; j < 2; ++j) {
            const int n  = bn + wn + j * 16 + l16;
            const float bv = bias[n];
            #pragma unroll
            for (int r = 0; r < 8; ++r) {
                const size_t idx = (size_t)(mBase + r) * N + n;
                const float v = acc[i][j][r] + bv;
                if constexpr (EPI == 0) {
                    outF[idx] = v;
                } else if constexpr (EPI == 1) {
                    outF[idx] = res[idx] + resScale * v;
                } else {
                    outH[idx] = (_Float16)(0.5f * v * (1.0f + erff(v * 0.70710678118654752f)));
                }
            }
        }
    }
}

// ---------------- row l2-normalize: wave per row -------------------------------------
__global__ void k_rownorm(const float* __restrict__ X, float* __restrict__ fn32,
                          _Float16* __restrict__ fn16)
{
    const int row  = blockIdx.x * 8 + (threadIdx.x >> 5);
    const int lane = threadIdx.x & 31;
    const float* x = X + (size_t)row * Dd;
    float s = 0.f;
    for (int d = lane; d < Dd; d += 32) { float v = x[d]; s += v * v; }
    for (int o = 16; o; o >>= 1) s += __shfl_xor(s, o, 32);
    const float inv = 1.0f / fmaxf(sqrtf(s), 1e-8f);
    for (int d = lane; d < Dd; d += 32) {
        const float v = x[d] * inv;
        fn32[(size_t)row * Dd + d] = v;
        fn16[(size_t)row * Dd + d] = (_Float16)v;
    }
}

// ---------------- weight convert+transpose f32[K,N] -> f16[N,K] -----------------------
__global__ void k_transpose16(const float* __restrict__ W, _Float16* __restrict__ WT,
                              int Kd, int Nn)
{
    const int idx = blockIdx.x * 256 + threadIdx.x;
    if (idx >= Kd * Nn) return;
    const int n = idx / Kd, k = idx - n * Kd;
    WT[(size_t)n * Kd + k] = (_Float16)W[(size_t)k * Nn + n];
}

// ---------------- anchor chain --------------------------------------------------------
__global__ void k_anchor_h1(const float* __restrict__ anchors, const float* __restrict__ w1,
                            const float* __restrict__ b1, float* __restrict__ h1)
{
    const int o = blockIdx.x * 256 + threadIdx.x;     // 10*512
    const int k = o >> 9, c = o & 511;
    const float* a = anchors + k * Dd;
    float s = b1[c];
    for (int d = 0; d < Dd; ++d) s += a[d] * w1[d * 512 + c];
    h1[o] = fmaxf(s, 0.0f);
}

__global__ void k_anchor_refine(const float* __restrict__ anchors, const float* __restrict__ h1,
                                const float* __restrict__ w2, const float* __restrict__ b2,
                                float* __restrict__ an)
{
    const int o = blockIdx.x * 256 + threadIdx.x;     // 10*1024
    const int k = o >> 10, d = o & 1023;
    const float* hr = h1 + k * 512;
    float s = b2[d];
    for (int c = 0; c < 512; ++c) s += hr[c] * w2[c * Dd + d];
    an[o] = anchors[o] + 0.1f * tanhf(s);
}

__global__ void k_anchor_norm(float* __restrict__ an, float* __restrict__ ansq)
{
    const int k = threadIdx.x >> 5, lane = threadIdx.x & 31;
    if (k >= Kk) return;
    float* a = an + k * Dd;
    float s = 0.f;
    for (int d = lane; d < Dd; d += 32) { float v = a[d]; s += v * v; }
    for (int o = 16; o; o >>= 1) s += __shfl_xor(s, o, 32);
    const float inv = 1.0f / fmaxf(sqrtf(s), 1e-8f);
    for (int d = lane; d < Dd; d += 32) a[d] *= inv;
    if (lane == 0) ansq[k] = s * inv * inv;
}

__global__ void k_kvproj(const float* __restrict__ an,
                         const float* __restrict__ kw, const float* __restrict__ kb,
                         const float* __restrict__ vw, const float* __restrict__ vb,
                         float* __restrict__ kp, float* __restrict__ vp)
{
    int o = blockIdx.x * 256 + threadIdx.x;           // 2*10*1024
    const float *W, *bb; float* out;
    if (o < Kk * Dd) { W = kw; bb = kb; out = kp; }
    else             { o -= Kk * Dd; W = vw; bb = vb; out = vp; }
    const int k = o >> 10, d = o & 1023;
    const float* a = an + k * Dd;
    float s = bb[d];
    for (int i = 0; i < Dd; ++i) s += a[i] * W[i * Dd + d];
    out[o] = s;
}

__global__ void k_div(const float* __restrict__ an, float* __restrict__ divOut)
{
    __shared__ float red[256];
    const int tid = threadIdx.x;
    float total = 0.f;
    for (int i = 0; i < Kk; ++i)
        for (int j = i + 1; j < Kk; ++j) {
            float p = 0.f;
            for (int d = tid; d < Dd; d += 256) {
                const float df = an[i * Dd + d] - an[j * Dd + d];
                p += df * df;
            }
            red[tid] = p; __syncthreads();
            for (int s = 128; s; s >>= 1) { if (tid < s) red[tid] += red[tid + s]; __syncthreads(); }
            if (tid == 0) total += sqrtf(fmaxf(red[0], 1e-24f));
            __syncthreads();
        }
    if (tid == 0) divOut[0] = expf(-fmaxf(total * (1.0f / 45.0f), 1e-6f));
}

// ---------------- attention (wave per (b,h), K=10) ------------------------------------
__global__ void k_attn(const float* __restrict__ q, const float* __restrict__ kp,
                       const float* __restrict__ vp, _Float16* __restrict__ att16,
                       float* __restrict__ avg_att)
{
    const int gw   = blockIdx.x * 8 + (threadIdx.x >> 5);
    const int lane = threadIdx.x & 31;
    const int b = gw >> 3, h = gw & 7;
    const float* qr = q + (size_t)b * Dd + h * 128;
    float qv[4];
    #pragma unroll
    for (int j = 0; j < 4; ++j) qv[j] = qr[lane * 4 + j];
    float sc[Kk], mx = -1e30f;
    #pragma unroll
    for (int k = 0; k < Kk; ++k) {
        const float* kr = kp + k * Dd + h * 128;
        float s = 0.f;
        #pragma unroll
        for (int j = 0; j < 4; ++j) s += qv[j] * kr[lane * 4 + j];
        for (int o = 16; o; o >>= 1) s += __shfl_xor(s, o, 32);
        s *= 0.088388347648318447f;   // 1/sqrt(128)
        sc[k] = s; mx = fmaxf(mx, s);
    }
    float den = 0.f;
    #pragma unroll
    for (int k = 0; k < Kk; ++k) { sc[k] = expf(sc[k] - mx); den += sc[k]; }
    const float rden = 1.0f / den;
    float o4[4] = {0.f, 0.f, 0.f, 0.f};
    #pragma unroll
    for (int k = 0; k < Kk; ++k) {
        const float aw = sc[k] * rden;
        const float* vr = vp + k * Dd + h * 128;
        #pragma unroll
        for (int j = 0; j < 4; ++j) o4[j] += aw * vr[lane * 4 + j];
        if (lane == 0) atomicAdd(&avg_att[(size_t)b * Kk + k], aw * 0.125f);
    }
    _Float16* ar = att16 + (size_t)b * Dd + h * 128;
    #pragma unroll
    for (int j = 0; j < 4; ++j) ar[lane * 4 + j] = (_Float16)o4[j];
}

// ---------------- layernorm (wave per row) -> f16 -------------------------------------
__global__ void k_layernorm(const float* __restrict__ enh, const float* __restrict__ g,
                            const float* __restrict__ bta, _Float16* __restrict__ x16)
{
    const int row  = blockIdx.x * 8 + (threadIdx.x >> 5);
    const int lane = threadIdx.x & 31;
    const float* e = enh + (size_t)row * Dd;
    float s = 0.f, s2 = 0.f;
    for (int d = lane; d < Dd; d += 32) { float v = e[d]; s += v; s2 += v * v; }
    for (int o = 16; o; o >>= 1) { s += __shfl_xor(s, o, 32); s2 += __shfl_xor(s2, o, 32); }
    const float mu  = s * (1.0f / Dd);
    const float var = s2 * (1.0f / Dd) - mu * mu;
    const float r   = rsqrtf(var + 1e-5f);
    for (int d = lane; d < Dd; d += 32)
        x16[(size_t)row * Dd + d] = (_Float16)((e[d] - mu) * r * g[d] + bta[d]);
}

// ---------------- center loss (wave per row) ------------------------------------------
__global__ void k_center(const float* __restrict__ fn, const float* __restrict__ an,
                         const float* __restrict__ ansq, const float* __restrict__ avg_att,
                         float* __restrict__ cacc)
{
    const int row  = blockIdx.x * 8 + (threadIdx.x >> 5);
    const int lane = threadIdx.x & 31;
    const float* f = fn + (size_t)row * Dd;
    float fv[32], sf = 0.f;
    #pragma unroll
    for (int j = 0; j < 32; ++j) { float v = f[lane + 32 * j]; fv[j] = v; sf += v * v; }
    for (int o = 16; o; o >>= 1) sf += __shfl_xor(sf, o, 32);
    float rs = 0.f;
    for (int k = 0; k < Kk; ++k) {
        const float* a = an + k * Dd;
        float d = 0.f;
        #pragma unroll
        for (int j = 0; j < 32; ++j) d += fv[j] * a[lane + 32 * j];
        for (int o = 16; o; o >>= 1) d += __shfl_xor(d, o, 32);
        const float sq = sf + ansq[k] - 2.0f * d;
        rs += avg_att[(size_t)row * Kk + k] * sqrtf(fmaxf(sq, 1e-12f));
    }
    if (lane == 0) atomicAdd(cacc, rs * (1.0f / (float)Bsz));
}

__global__ void k_geo(const float* __restrict__ scal, const float* __restrict__ div_w,
                      const float* __restrict__ cen_w, float* __restrict__ outGeo)
{
    const float g = fabsf(div_w[0]) * scal[0] + fabsf(cen_w[0]) * scal[1];
    outGeo[0] = fminf(fmaxf(g, 0.0f), 0.01f);
}

// ---------------- workspace layout ----------------------------------------------------
static constexpr size_t SZ_F16  = (size_t)Bsz * Dd * 2;     // 32 MB
static constexpr size_t SZ_F32  = (size_t)Bsz * Dd * 4;     // 64 MB
static constexpr size_t SZ_W    = (size_t)Dd * Dd * 2;      // 2 MB
static constexpr size_t OFF_FN16 = 0;                        // fn16, later x16
static constexpr size_t OFF_BUFB = OFF_FN16 + SZ_F16;        // q32 -> enh32 -> h16
static constexpr size_t OFF_ATT  = OFF_BUFB + SZ_F32;        // att16
static constexpr size_t OFF_QWT  = OFF_ATT + SZ_F16;
static constexpr size_t OFF_OWT  = OFF_QWT + SZ_W;
static constexpr size_t OFF_FF1T = OFF_OWT + SZ_W;
static constexpr size_t OFF_FF2T = OFF_FF1T + 2 * SZ_W;
static constexpr size_t OFF_H1   = OFF_FF2T + 2 * SZ_W;
static constexpr size_t OFF_AN   = OFF_H1 + (size_t)Kk * 512 * 4;
static constexpr size_t OFF_KP   = OFF_AN + (size_t)Kk * Dd * 4;
static constexpr size_t OFF_VP   = OFF_KP + (size_t)Kk * Dd * 4;
static constexpr size_t OFF_ANSQ = OFF_VP + (size_t)Kk * Dd * 4;
static constexpr size_t OFF_AVG  = OFF_ANSQ + 256;
static constexpr size_t OFF_SCAL = OFF_AVG + (size_t)Bsz * Kk * 4;

extern "C" void kernel_launch(void* const* d_in, const int* in_sizes, int n_in,
                              void* d_out, int out_size, void* d_ws, size_t ws_size,
                              hipStream_t stream)
{
    const float* features = (const float*)d_in[0];
    const float* anchors  = (const float*)d_in[1];
    const float* ar_w1 = (const float*)d_in[2];  const float* ar_b1 = (const float*)d_in[3];
    const float* ar_w2 = (const float*)d_in[4];  const float* ar_b2 = (const float*)d_in[5];
    const float* q_w   = (const float*)d_in[6];  const float* q_b   = (const float*)d_in[7];
    const float* k_w   = (const float*)d_in[8];  const float* k_b   = (const float*)d_in[9];
    const float* v_w   = (const float*)d_in[10]; const float* v_b   = (const float*)d_in[11];
    const float* o_w   = (const float*)d_in[12]; const float* o_b   = (const float*)d_in[13];
    const float* ln_g  = (const float*)d_in[14]; const float* ln_b  = (const float*)d_in[15];
    const float* ff_w1 = (const float*)d_in[16]; const float* ff_b1 = (const float*)d_in[17];
    const float* ff_w2 = (const float*)d_in[18]; const float* ff_b2 = (const float*)d_in[19];
    const float* div_w = (const float*)d_in[20]; const float* cen_w = (const float*)d_in[21];

    char* ws = (char*)d_ws;
    _Float16* fn16  = (_Float16*)(ws + OFF_FN16);
    _Float16* x16   = fn16;                          // reuse after fn16 consumed
    float*    q32   = (float*)(ws + OFF_BUFB);
    float*    enh32 = q32;                           // reuse after q consumed
    _Float16* h16   = (_Float16*)(ws + OFF_BUFB);    // reuse after enh consumed
    _Float16* att16 = (_Float16*)(ws + OFF_ATT);
    _Float16* qwT   = (_Float16*)(ws + OFF_QWT);
    _Float16* owT   = (_Float16*)(ws + OFF_OWT);
    _Float16* ff1T  = (_Float16*)(ws + OFF_FF1T);
    _Float16* ff2T  = (_Float16*)(ws + OFF_FF2T);
    float*    h1    = (float*)(ws + OFF_H1);
    float*    an    = (float*)(ws + OFF_AN);
    float*    kp    = (float*)(ws + OFF_KP);
    float*    vp    = (float*)(ws + OFF_VP);
    float*    ansq  = (float*)(ws + OFF_ANSQ);
    float*    avg   = (float*)(ws + OFF_AVG);
    float*    scal  = (float*)(ws + OFF_SCAL);       // [0]=div_loss, [1]=center_sum

    float* outFinal = (float*)d_out;
    float* outGeo   = outFinal + (size_t)Bsz * Dd;

    // zero atomic accumulators (graph-capture safe)
    (void)hipMemsetAsync(ws + OFF_AVG, 0, (size_t)Bsz * Kk * 4 + 256, stream);

    // fn = l2norm(features): fn32 -> d_out, fn16 -> ws
    k_rownorm<<<Bsz / 8, 256, 0, stream>>>(features, outFinal, fn16);

    // weight convert/transpose to f16 [N,K]
    k_transpose16<<<(Dd * Dd + 255) / 256, 256, 0, stream>>>(q_w,  qwT,  Dd, Dd);
    k_transpose16<<<(Dd * Dd + 255) / 256, 256, 0, stream>>>(o_w,  owT,  Dd, Dd);
    k_transpose16<<<(Dd * 2 * Dd + 255) / 256, 256, 0, stream>>>(ff_w1, ff1T, Dd, 2 * Dd);
    k_transpose16<<<(2 * Dd * Dd + 255) / 256, 256, 0, stream>>>(ff_w2, ff2T, 2 * Dd, Dd);

    // anchor chain
    k_anchor_h1<<<(Kk * 512) / 256, 256, 0, stream>>>(anchors, ar_w1, ar_b1, h1);
    k_anchor_refine<<<(Kk * Dd) / 256, 256, 0, stream>>>(anchors, h1, ar_w2, ar_b2, an);
    k_anchor_norm<<<1, 320, 0, stream>>>(an, ansq);
    k_kvproj<<<(2 * Kk * Dd) / 256, 256, 0, stream>>>(an, k_w, k_b, v_w, v_b, kp, vp);
    k_div<<<1, 256, 0, stream>>>(an, &scal[0]);

    // G1: q = fn @ q_w + q_b   (f32 out)
    wmma_gemm_epi<0><<<dim3(Bsz / 128, Dd / 128), 256, 0, stream>>>(
        fn16, qwT, q_b, nullptr, q32, nullptr, Bsz, Dd, Dd, 0.f);

    // attention: scores/softmax/att + avg_att
    k_attn<<<(Bsz * Hh) / 8, 256, 0, stream>>>(q32, kp, vp, att16, avg);

    // G2: enh = fn + 0.1*(att @ o_w + o_b)
    wmma_gemm_epi<1><<<dim3(Bsz / 128, Dd / 128), 256, 0, stream>>>(
        att16, owT, o_b, outFinal, enh32, nullptr, Bsz, Dd, Dd, 0.1f);

    // layernorm(enh) -> x16
    k_layernorm<<<Bsz / 8, 256, 0, stream>>>(enh32, ln_g, ln_b, x16);

    // center loss (reads fn32 from d_out before the final in-place epilogue)
    k_center<<<Bsz / 8, 256, 0, stream>>>(outFinal, an, ansq, avg, &scal[1]);

    // G3: h = gelu(x @ ff_w1 + b1) -> f16
    wmma_gemm_epi<2><<<dim3(Bsz / 128, 2 * Dd / 128), 256, 0, stream>>>(
        x16, ff1T, ff_b1, nullptr, nullptr, h16, Bsz, 2 * Dd, Dd, 0.f);

    // G4: final = fn + 0.2*(h @ ff_w2 + b2)   (in-place on d_out)
    wmma_gemm_epi<1><<<dim3(Bsz / 128, Dd / 128), 256, 0, stream>>>(
        h16, ff2T, ff_b2, outFinal, outFinal, nullptr, Bsz, Dd, 2 * Dd, 0.2f);

    // geo scalar
    k_geo<<<1, 1, 0, stream>>>(scal, div_w, cen_w, outGeo);

    (void)in_sizes; (void)n_in; (void)out_size; (void)ws_size;
}
